// SparseLossFeature_60395830117050
// MI455X (gfx1250) — matrix operand order repaired and checked by reading, not compiled, not run on hardware
//
#include <hip/hip_runtime.h>
#include <hip/hip_bf16.h>

// ---------------------------------------------------------------------------
// SparseLossFeature on MI455X (gfx1250):
//   conv(512 filters x 16384 taps over 65536 samples, 2 batches) as an
//   implicit GEMM in bf16 WMMA (16x16x32, f32 accum).
//   - A fragments: two ds_load_b128 per lane (this code's K order: lane half hf
//     holds contiguous K = 16*hf + 0..15, same as B).
//   - B (Toeplitz) fragments: two global b128 loads per lane from a
//     pre-reversed x buffer (reversal makes each lane's window forward-
//     contiguous; window is L0-resident).
//   - A tiles staged LDS via TDM tensor_load_to_lds, double-buffered,
//     fenced with s_wait_tensorcnt (fallback: uint4 staging).
//   Fused relu+dropout+sum epilogue; exact top-512 via bf16-key histogram,
//   bitonic sort, projection gather.
// ---------------------------------------------------------------------------

typedef __attribute__((ext_vector_type(16))) __bf16 v16bf;
typedef __attribute__((ext_vector_type(8)))  float  v8f;
typedef unsigned int u32x4 __attribute__((ext_vector_type(4)));
typedef int          i32x4 __attribute__((ext_vector_type(4)));
typedef int          i32x8 __attribute__((ext_vector_type(8)));

#define N_BATCH   2
#define N_CH      512
#define N_T       65536
#define N_TAPS    16384
#define FLAT_PER_B 33554432ull          // 512*65536 = 2^25
#define XPAD_LEN  (N_TAPS + N_T)        // 81920
#define TILE_T    128
#define TILE_C    128
#define KSTEP     32
#define N_ITER    (N_TAPS / KSTEP)      // 512
#define N_KEEP    512
#define DROP_P    0.05f
#define KEEP_SCL  (1.0f / 0.95f)
#define A_TILE_BYTES (TILE_C * KSTEP * 2)   // 8192 B per buffer

#if defined(__has_builtin)
#if __has_builtin(__builtin_amdgcn_tensor_load_to_lds)
#define HAVE_TDM 1
#endif
#endif

static __device__ __forceinline__ unsigned short f2bfbits(float f) {
  unsigned u; __builtin_memcpy(&u, &f, 4);
  unsigned r = u + 0x7FFFu + ((u >> 16) & 1u);   // round-to-nearest-even
  return (unsigned short)(r >> 16);
}
static __device__ __forceinline__ float bfbits2f(unsigned short h) {
  unsigned u = ((unsigned)h) << 16;
  float f; __builtin_memcpy(&f, &u, 4);
  return f;
}

#ifdef HAVE_TDM
// 2D TDM descriptor: tile (KSTEP x TILE_C) of 2-byte elements from the
// (512 x 16384) filter tensor, row stride 16384 elements, into LDS.
static __device__ __forceinline__ void tdm_load_a_tile(unsigned ldsOff,
                                                       const unsigned short* gsrc) {
  const unsigned long long ga = (unsigned long long)(size_t)gsrc;
  u32x4 g0;
  g0[0] = 1u;                                   // count=1, user mode
  g0[1] = ldsOff;                               // lds_addr (sA is first LDS var)
  g0[2] = (unsigned)(ga & 0xFFFFFFFFull);       // global_addr[31:0]
  g0[3] = (unsigned)((ga >> 32) & 0x01FFFFFFull) | (2u << 30);  // addr hi | type=2
  i32x8 g1 = { 0x00010000,                      // data_size=1 (2 B elements)
               0x40000000,                      // tensor_dim0 = 16384 (lo16<<16)
               0x02000000,                      // tensor_dim1 = 512   (lo16<<16)
               0x00200000,                      // tile_dim0   = 32    (<<16)
               TILE_C,                          // tile_dim1   = 128
               N_TAPS,                          // tensor_dim0_stride = 16384
               0, 0 };
  i32x4 z4 = { 0, 0, 0, 0 };
#if __clang_major__ >= 23
  i32x8 z8 = { 0, 0, 0, 0, 0, 0, 0, 0 };
  __builtin_amdgcn_tensor_load_to_lds(g0, g1, z4, z4, z8, 0);
#else
  __builtin_amdgcn_tensor_load_to_lds(g0, g1, z4, z4, 0);
#endif
}
#endif

// ---------------------------------------------------------------------------
// Prep kernels
// ---------------------------------------------------------------------------
__global__ void zero_u32_kernel(unsigned* p, int n) {
  int i = blockIdx.x * blockDim.x + threadIdx.x;
  for (; i < n; i += gridDim.x * blockDim.x) p[i] = 0u;
}

__global__ void cvt_filters_kernel(const float* __restrict__ f,
                                   unsigned short* __restrict__ fbf, int n) {
  int i = blockIdx.x * blockDim.x + threadIdx.x;
  for (; i < n; i += gridDim.x * blockDim.x) fbf[i] = f2bfbits(f[i]);
}

// Reversed, zero-padded x in bf16: xrev[b][m] = x[b][65535-m] for m<=65535,
// else 0 (the 16384-tap history before t=0).
__global__ void cvt_xrev_kernel(const float* __restrict__ x,
                                unsigned short* __restrict__ xrev) {
  int i = blockIdx.x * blockDim.x + threadIdx.x;     // covers 2*81920 exactly
  int b = i / XPAD_LEN;
  int m = i - b * XPAD_LEN;
  float v = (m < N_T) ? x[b * N_T + (N_T - 1 - m)] : 0.0f;
  xrev[i] = f2bfbits(v);
}

// ---------------------------------------------------------------------------
// Implicit-GEMM conv: res[b,c,t] = sum_tau f[c,tau] * x[b,t-tau]
//   grid = (N_T/128, N_CH/128, N_BATCH), 256 threads = 8 wave32.
//   wave (mw,nw): 32 channels x 64 times -> 2 A frags x 4 B frags = 8 WMMA.
// ---------------------------------------------------------------------------
__global__ __launch_bounds__(256) void conv_wmma_kernel(
    const unsigned short* __restrict__ fbf,     // (512, 16384) bf16 bits
    const unsigned short* __restrict__ xrev,    // (2, 81920) reversed bf16
    const float* __restrict__ U,                // (2, 2^25) dropout uniforms
    unsigned short* __restrict__ resbf,         // (2, 2^25) bf16 bits out
    float* __restrict__ partial)                // (4096,) per-block sums
{
  const int tid  = threadIdx.x;
  const int lane = tid & 31;
  const int wave = tid >> 5;
  const int mw   = wave >> 1;        // 0..3 : 32-channel sub-tile
  const int nw   = wave & 1;         // 0..1 : 64-time sub-tile
  const int hf   = lane >> 4;        // half-wave (ISA fragment layout)
  const int l16  = lane & 15;

  const int tBase = blockIdx.x * TILE_T;
  const int cBase = blockIdx.y * TILE_C;
  const int b     = blockIdx.z;

  // sA declared first -> LDS offset 0 (TDM descriptor assumes this).
  __shared__ __align__(16) unsigned short sA[2][TILE_C][KSTEP];  // 16 KB
  __shared__ float sRed[256];

  v8f acc[2][4] = {};
  const unsigned short* xr = xrev + (size_t)b * XPAD_LEN;

#ifdef HAVE_TDM
  if (tid < 32) tdm_load_a_tile(0u, fbf + (size_t)cBase * N_TAPS);
#endif

  for (int it = 0; it < N_ITER; ++it) {
    const int k = it * KSTEP;
#ifdef HAVE_TDM
    const int cur = it & 1;
    if (tid < 32) __builtin_amdgcn_s_wait_tensorcnt(0);
    __syncthreads();                                   // tile `cur` resident
    if (tid < 32 && it + 1 < N_ITER)                   // prefetch next tile
      tdm_load_a_tile((unsigned)(((it + 1) & 1) * A_TILE_BYTES),
                      fbf + (size_t)cBase * N_TAPS + (size_t)(k + KSTEP));
#else
    const int cur = 0;
    __syncthreads();
    {   // 256 threads x 2 x uint4 = 8 KB tile
      const int row  = tid >> 2;         // 0..63
      const int part = tid & 3;          // 16-byte chunk
      const size_t s0 = (size_t)(cBase + row) * N_TAPS + k + part * 8;
      *reinterpret_cast<uint4*>(&sA[0][row][part * 8]) =
          *reinterpret_cast<const uint4*>(&fbf[s0]);
      const size_t s1 = (size_t)(cBase + 64 + row) * N_TAPS + k + part * 8;
      *reinterpret_cast<uint4*>(&sA[0][64 + row][part * 8]) =
          *reinterpret_cast<const uint4*>(&fbf[s1]);
      if (k + KSTEP < N_TAPS) __builtin_prefetch(&fbf[s0 + KSTEP], 0, 1);
    }
    __syncthreads();
#endif

    // A fragments: per lane two contiguous aligned 16 B runs -> ds_load_b128.
    union AU { uint4 q[2]; v16bf v; } au[2];
#pragma unroll
    for (int m = 0; m < 2; ++m) {
      const uint4* aRow = reinterpret_cast<const uint4*>(
          &sA[cur][(mw << 5) + (m << 4) + l16][0]);
      au[m].q[0] = aRow[2 * hf];       // K = 16*hf + 0..7
      au[m].q[1] = aRow[2 * hf + 1];   // K = 16*hf + 8..15
    }

    // B fragments from reversed x: lane run xr[65535 + k - tBase - tl + 16*hf + j]
    const long long xbase = 65535LL + k - tBase + (hf << 4);
#pragma unroll
    for (int g = 0; g < 4; ++g) {
      const int tl = (nw << 6) + (g << 4) + l16;
      const uint4* bp = reinterpret_cast<const uint4*>(xr + (xbase - tl));
      union BU { uint4 q[2]; v16bf v; } bu;
      bu.q[0] = bp[0];
      bu.q[1] = bp[1];
#pragma unroll
      for (int m = 0; m < 2; ++m)
        acc[m][g] = __builtin_amdgcn_wmma_f32_16x16x32_bf16(
            false, au[m].v, false, bu.v, (short)0, acc[m][g], false, false);
    }
  }

  // Epilogue: relu + inverted dropout + bf16 store + local sum.
  float lsum = 0.0f;
  const unsigned long long base = (unsigned long long)b * FLAT_PER_B;
#pragma unroll
  for (int m = 0; m < 2; ++m) {
#pragma unroll
    for (int g = 0; g < 4; ++g) {
      const int t = tBase + (nw << 6) + (g << 4) + l16;
#pragma unroll
      for (int r = 0; r < 8; ++r) {
        const int ch = cBase + (mw << 5) + (m << 4) + (hf << 3) + r;
        float v = acc[m][g][r];
        v = v > 0.0f ? v : 0.0f;
        const unsigned long long flat =
            ((unsigned long long)ch << 16) | (unsigned)t;
        const float u = U[base + flat];
        v *= (u >= DROP_P) ? KEEP_SCL : 0.0f;
        resbf[base + flat] = f2bfbits(v);
        lsum += v;
      }
    }
  }

  // Deterministic tree reduction -> per-block partial.
  sRed[tid] = lsum;
  __syncthreads();
  for (int s = 128; s > 0; s >>= 1) {
    if (tid < s) sRed[tid] += sRed[tid + s];
    __syncthreads();
  }
  if (tid == 0)
    partial[((unsigned)blockIdx.z * 4u + blockIdx.y) * 512u + blockIdx.x] = sRed[0];
}

// ---------------------------------------------------------------------------
// Deterministic reduction of 2048 partials per batch.
// ---------------------------------------------------------------------------
__global__ __launch_bounds__(256) void reduce_sum_kernel(
    const float* __restrict__ partial, float* __restrict__ sums) {
  __shared__ float s[256];
  const int b = blockIdx.x;
  float acc = 0.0f;
  for (int i = threadIdx.x; i < 2048; i += 256) acc += partial[b * 2048 + i];
  s[threadIdx.x] = acc;
  __syncthreads();
  for (int st = 128; st > 0; st >>= 1) {
    if (threadIdx.x < st) s[threadIdx.x] += s[threadIdx.x + st];
    __syncthreads();
  }
  if (threadIdx.x == 0) sums[b] = s[0];
}

// ---------------------------------------------------------------------------
// Top-k machinery: bf16-key histogram (values >= 0 -> 15-bit monotone key).
// ---------------------------------------------------------------------------
__global__ void hist_kernel(const unsigned short* __restrict__ resbf,
                            unsigned* __restrict__ hist) {
  unsigned long long i = (unsigned long long)blockIdx.x * blockDim.x + threadIdx.x;
  const unsigned long long stride = (unsigned long long)gridDim.x * blockDim.x;
  const unsigned long long total = (unsigned long long)N_BATCH * FLAT_PER_B;
  for (; i < total; i += stride) {
    const unsigned b   = (unsigned)(i >> 25);
    const unsigned key = resbf[i] & 0x7FFFu;
    if (i + stride < total) __builtin_prefetch(&resbf[i + stride], 0, 0);
    atomicAdd(&hist[b * 32768u + key], 1u);
  }
}

__global__ void thresh_kernel(const unsigned* __restrict__ hist,
                              int* __restrict__ thresh, int* __restrict__ ngt) {
  if (threadIdx.x != 0) return;
  const int b = blockIdx.x;
  const unsigned* h = hist + b * 32768;
  unsigned acc = 0;
  for (int v = 32767; v >= 0; --v) {
    const unsigned c = h[v];
    if (acc + c >= (unsigned)N_KEEP) { thresh[b] = v; ngt[b] = (int)acc; return; }
    acc += c;
  }
  thresh[b] = 0;
  ngt[b] = (int)acc;
}

__global__ void compact_kernel(const unsigned short* __restrict__ resbf,
                               const int* __restrict__ thresh,
                               const int* __restrict__ ngt,
                               unsigned* __restrict__ ctrGT,
                               unsigned* __restrict__ ctrEQ,
                               float* __restrict__ candV,
                               int* __restrict__ candI) {
  unsigned long long i = (unsigned long long)blockIdx.x * blockDim.x + threadIdx.x;
  const unsigned long long stride = (unsigned long long)gridDim.x * blockDim.x;
  const unsigned long long total = (unsigned long long)N_BATCH * FLAT_PER_B;
  for (; i < total; i += stride) {
    const unsigned b   = (unsigned)(i >> 25);
    const unsigned fi  = (unsigned)(i & (FLAT_PER_B - 1ull));
    const int key = (int)(resbf[i] & 0x7FFFu);
    const int th  = thresh[b];
    if (key > th) {
      const unsigned p = atomicAdd(&ctrGT[b], 1u);
      if (p < (unsigned)N_KEEP) {
        candV[b * N_KEEP + p] = bfbits2f((unsigned short)key);
        candI[b * N_KEEP + p] = (int)fi;
      }
    } else if (key == th) {
      const int navail = N_KEEP - ngt[b];
      const unsigned p = atomicAdd(&ctrEQ[b], 1u);
      if ((int)p < navail) {
        const int slot = ngt[b] + (int)p;
        candV[b * N_KEEP + slot] = bfbits2f((unsigned short)key);
        candI[b * N_KEEP + slot] = (int)fi;
      }
    }
  }
}

// ---------------------------------------------------------------------------
// Sort 512 candidates (bitonic, desc value / asc index), normalize, gather.
// ---------------------------------------------------------------------------
__global__ __launch_bounds__(512) void topk_emit_kernel(
    const float* __restrict__ candV, const int* __restrict__ candI,
    const float* __restrict__ sums,
    const float* __restrict__ time_proj,      // (65536, 16)
    const float* __restrict__ channel_proj,   // (512, 16)
    float* __restrict__ out)                  // (2, 512, 32)
{
  __shared__ float sv[N_KEEP];
  __shared__ int   si[N_KEEP];
  const int b = blockIdx.x;
  const int i = threadIdx.x;
  sv[i] = candV[b * N_KEEP + i];
  si[i] = candI[b * N_KEEP + i];
  __syncthreads();

  for (int k2 = 2; k2 <= N_KEEP; k2 <<= 1) {
    for (int j = k2 >> 1; j > 0; j >>= 1) {
      const int ixj = i ^ j;
      if (ixj > i) {
        const bool  dir = ((i & k2) == 0);
        const float v0 = sv[i], v1 = sv[ixj];
        const int   i0 = si[i], i1 = si[ixj];
        const bool before = (v0 > v1) || (v0 == v1 && i0 < i1);
        if (before != dir) { sv[i] = v1; si[i] = i1; sv[ixj] = v0; si[ixj] = i0; }
      }
      __syncthreads();
    }
  }

  const float norm = sv[i] / sums[b];
  const int idx  = si[i];
  const int tIdx = idx & (N_T - 1);
  const int aIdx = idx >> 16;
  float* o = out + ((size_t)b * N_KEEP + i) * 32;
#pragma unroll
  for (int j = 0; j < 16; ++j) o[j]      = time_proj[tIdx * 16 + j] * norm;
#pragma unroll
  for (int j = 0; j < 16; ++j) o[16 + j] = channel_proj[aIdx * 16 + j] * norm;
}

// ---------------------------------------------------------------------------
// Host-side orchestration
// ---------------------------------------------------------------------------
extern "C" void kernel_launch(void* const* d_in, const int* in_sizes, int n_in,
                              void* d_out, int out_size, void* d_ws, size_t ws_size,
                              hipStream_t stream) {
  const float* x            = (const float*)d_in[0];  // (2,1,65536)
  const float* filters      = (const float*)d_in[1];  // (512,16384)
  const float* time_proj    = (const float*)d_in[2];  // (65536,16)
  const float* channel_proj = (const float*)d_in[3];  // (512,16)
  const float* dropout_u    = (const float*)d_in[4];  // (2, 2^25)
  float* out = (float*)d_out;                         // (2,512,32)
  char*  ws  = (char*)d_ws;

  // Workspace layout (256 B aligned regions)
  const size_t OFF_FBF  = 0;                                       // 16 MB
  const size_t OFF_XREV = OFF_FBF  + (size_t)N_CH * N_TAPS * 2;    // 320 KB
  const size_t OFF_RES  = OFF_XREV + (size_t)N_BATCH * XPAD_LEN * 2; // 134 MB
  const size_t OFF_PART = OFF_RES  + (size_t)N_BATCH * FLAT_PER_B * 2;
  const size_t OFF_SUM  = OFF_PART + 4096 * sizeof(float);
  const size_t OFF_HIST = OFF_SUM  + 256;                          // 256 KB
  const size_t OFF_CTR  = OFF_HIST + 2 * 32768 * sizeof(unsigned); // contiguous
  const size_t OFF_CANDV = OFF_CTR + 256;
  const size_t OFF_CANDI = OFF_CANDV + N_BATCH * N_KEEP * sizeof(float);

  unsigned short* fbf     = (unsigned short*)(ws + OFF_FBF);
  unsigned short* xrev    = (unsigned short*)(ws + OFF_XREV);
  unsigned short* resbf   = (unsigned short*)(ws + OFF_RES);
  float*          partial = (float*)(ws + OFF_PART);
  float*          sums    = (float*)(ws + OFF_SUM);
  unsigned*       hist    = (unsigned*)(ws + OFF_HIST);
  unsigned*       ctrGT   = (unsigned*)(ws + OFF_CTR);
  unsigned*       ctrEQ   = ctrGT + 2;
  int*            thresh  = (int*)(ctrGT + 4);
  int*            ngt     = (int*)(ctrGT + 6);
  float*          candV   = (float*)(ws + OFF_CANDV);
  int*            candI   = (int*)(ws + OFF_CANDI);

  // 1) zero hist + counters (hist and ctr zone contiguous: 65536+8 u32)
  zero_u32_kernel<<<256, 256, 0, stream>>>(hist, 65536 + 8);

  // 2) precision conversion + x reversal
  cvt_filters_kernel<<<4096, 256, 0, stream>>>(filters, fbf, N_CH * N_TAPS);
  cvt_xrev_kernel<<<(N_BATCH * XPAD_LEN) / 256, 256, 0, stream>>>(x, xrev);

  // 3) conv as implicit GEMM (bf16 WMMA, TDM-staged A tiles)
  dim3 grid(N_T / TILE_T, N_CH / TILE_C, N_BATCH);   // (512, 4, 2)
  conv_wmma_kernel<<<grid, 256, 0, stream>>>(fbf, xrev, dropout_u, resbf, partial);

  // 4) deterministic total sums
  reduce_sum_kernel<<<N_BATCH, 256, 0, stream>>>(partial, sums);

  // 5) exact top-512 select via bf16-key histogram
  hist_kernel<<<8192, 256, 0, stream>>>(resbf, hist);
  thresh_kernel<<<N_BATCH, 64, 0, stream>>>(hist, thresh, ngt);
  compact_kernel<<<8192, 256, 0, stream>>>(resbf, thresh, ngt, ctrGT, ctrEQ,
                                           candV, candI);

  // 6) sort + normalize + projection gather
  topk_emit_kernel<<<N_BATCH, N_KEEP, 0, stream>>>(candV, candI, sums,
                                                   time_proj, channel_proj, out);
}
